// EarlyFusionMamba_3238405341868
// MI455X (gfx1250) — compile-verified, matrix operand order checked by reading, not run on hardware
//
#include <hip/hip_runtime.h>

#define DEVFN __device__ __forceinline__

typedef __bf16 bf16_t;
typedef __attribute__((ext_vector_type(16))) __bf16 v16bf;
typedef __attribute__((ext_vector_type(8)))  __bf16 v8bf;
typedef __attribute__((ext_vector_type(8)))  float  v8f;

union BFrag { v16bf v; v8bf h[2]; };

DEVFN v8f wmma_bf16(v16bf a, v16bf b, v8f c) {
  // (neg_a, A, neg_b, B, c_mod, C, reuse_a, reuse_b)
  return __builtin_amdgcn_wmma_f32_16x16x32_bf16(false, a, false, b, (short)0, c, false, false);
}

DEVFN float sigmoidf_(float x) { return 1.0f / (1.0f + __expf(-x)); }
DEVFN float siluf_(float x)    { return x * sigmoidf_(x); }
DEVFN float softplusf_(float x){ return (x > 20.0f) ? x : log1pf(__expf(x)); }

DEVFN void async_b128_to_lds(unsigned ldsOff, unsigned long long gaddr) {
  // GLOBAL_LOAD_ASYNC_TO_LDS_B128: vdst = LDS byte offset VGPR, vaddr = 64-bit addr
  asm volatile("global_load_async_to_lds_b128 %0, %1, off"
               :: "v"(ldsOff), "v"(gaddr) : "memory");
}
DEVFN void wait_asynccnt0() {
  asm volatile("s_wait_asynccnt 0x0" ::: "memory");
}

// ---------------------------------------------------------------------------
// bf16 WMMA GEMM: C(M,N) = A(M,K) * W(N,K)^T  [+bias per col][+ACT]
// block = 128 threads (4 waves), launch_bounds(128,1) so the scheduler may
// keep all operand fragments live. Block tile = 128 rows x 64 cols.
// Wave computes 32x64 (2x4 WMMA accumulators). The 64x32 W tile is staged in
// LDS via global_load_async_to_lds_b128 (double-buffered, ASYNCcnt pipelined);
// A fragments stream from global (L2-resident for these shapes). All 4 B
// fragments are preloaded before the WMMA burst so LDS latency is covered by
// partial dscnt waits instead of 4 full stalls per K-step.
// Requires: M % 128 == 0, N % 64 == 0, K % 32 == 0 (true for all our shapes).
// ACT: 0 = none, 1 = softplus
// ---------------------------------------------------------------------------
template <int ACT>
__global__ __launch_bounds__(128, 1)
void gemm_bf16_wmma(const bf16_t* __restrict__ A, const bf16_t* __restrict__ W,
                    float* __restrict__ C, const float* __restrict__ bias,
                    int M, int N, int K, int ldc)
{
  __shared__ bf16_t tW[2][64 * 32];   // double-buffered 4KB W tiles

  const int tid  = threadIdx.x;
  const int lane = tid & 31;
  const int wave = tid >> 5;
  const int n0 = blockIdx.x * 64;
  const int m0 = (blockIdx.y * 4 + wave) * 32;

  const int r  = lane & 15;
  const int kb = (lane >> 4) * 8;   // lanes 0-15: K 0..7/16..23; 16-31: 8..15/24..31

  const bf16_t* Ar0 = A + (size_t)(m0 + r)      * K;
  const bf16_t* Ar1 = A + (size_t)(m0 + 16 + r) * K;

  // W tile = 64 rows x 32 k (bf16) = 256 x 16B chunks; each thread DMAs 2.
  const int ch0 = tid, ch1 = tid + 128;
  const int row0 = ch0 >> 2, ko0 = (ch0 & 3) * 8;
  const int row1 = ch1 >> 2, ko1 = (ch1 & 3) * 8;

  auto issueW = [&](int buf, int k) {
    async_b128_to_lds((unsigned)(uintptr_t)&tW[buf][row0 * 32 + ko0],
                      (unsigned long long)(W + (size_t)(n0 + row0) * K + k + ko0));
    async_b128_to_lds((unsigned)(uintptr_t)&tW[buf][row1 * 32 + ko1],
                      (unsigned long long)(W + (size_t)(n0 + row1) * K + k + ko1));
  };

  v8f acc[2][4];
#pragma unroll
  for (int i = 0; i < 2; ++i)
#pragma unroll
    for (int j = 0; j < 4; ++j) acc[i][j] = (v8f){};

  issueW(0, 0);
  wait_asynccnt0();
  __syncthreads();

  int cur = 0;
  for (int k = 0; k < K; k += 32) {
    const bool more = (k + 32) < K;
    if (more) issueW(cur ^ 1, k + 32);   // DMA next tile while computing this one

    BFrag a0, a1;
    a0.h[0] = *(const v8bf*)(Ar0 + k + kb);
    a0.h[1] = *(const v8bf*)(Ar0 + k + 16 + kb);
    a1.h[0] = *(const v8bf*)(Ar1 + k + kb);
    a1.h[1] = *(const v8bf*)(Ar1 + k + 16 + kb);

    // Preload all 4 B fragments (8x ds_load_b128) before any WMMA so waits
    // can be partial and LDS latency overlaps with matrix math.
    BFrag b[4];
#pragma unroll
    for (int nt = 0; nt < 4; ++nt) {
      const bf16_t* wsrc = &tW[cur][(nt * 16 + r) * 32 + kb];
      b[nt].h[0] = *(const v8bf*)(wsrc);
      b[nt].h[1] = *(const v8bf*)(wsrc + 16);
    }

#pragma unroll
    for (int nt = 0; nt < 4; ++nt) {
      acc[0][nt] = wmma_bf16(a0.v, b[nt].v, acc[0][nt]);
      acc[1][nt] = wmma_bf16(a1.v, b[nt].v, acc[1][nt]);
    }

    if (more) {
      wait_asynccnt0();   // our 2 DMAs for next tile have landed
      __syncthreads();    // everyone's have landed; also fences buffer reuse
    }
    cur ^= 1;
  }

  const int colOff = lane & 15;
  const int rowOff = (lane >> 4) * 8;
#pragma unroll
  for (int i = 0; i < 2; ++i) {
#pragma unroll
    for (int nt = 0; nt < 4; ++nt) {
      const int col = n0 + nt * 16 + colOff;
      const float bv = bias ? bias[col] : 0.0f;
      const int row = m0 + i * 16 + rowOff;
#pragma unroll
      for (int e = 0; e < 8; ++e) {
        float v = acc[i][nt][e] + bv;
        if (ACT == 1) v = softplusf_(v);
        C[(size_t)(row + e) * ldc + col] = v;
      }
    }
  }
}

// ---------------------------------------------------------------------------
// f32 -> bf16 cast (contiguous)
// ---------------------------------------------------------------------------
__global__ __launch_bounds__(256)
void cast_f32_bf16_kernel(const float* __restrict__ in, bf16_t* __restrict__ out, long n)
{
  long i = (long)blockIdx.x * 256 + threadIdx.x;
  if (i < n) out[i] = (bf16_t)in[i];
}

// f32 (rows, ld) take first `width` cols -> bf16 (rows, width)
__global__ __launch_bounds__(256)
void cast_strided_kernel(const float* __restrict__ in, int ld, int width,
                         bf16_t* __restrict__ out, long n)
{
  long i = (long)blockIdx.x * 256 + threadIdx.x;
  if (i >= n) return;
  long row = i / width;
  int  col = (int)(i - row * width);
  out[i] = (bf16_t)in[row * ld + col];
}

// concat(xv, broadcast(xi)) -> bf16 (M, 128)
__global__ __launch_bounds__(256)
void concat_cast_kernel(const float* __restrict__ xv, const float* __restrict__ xi,
                        bf16_t* __restrict__ out, long n, int Tlen)
{
  long i = (long)blockIdx.x * 256 + threadIdx.x;
  if (i >= n) return;
  int col = (int)(i & 127);
  long row = i >> 7;
  int b = (int)(row / Tlen);
  float v = (col < 96) ? xv[row * 96 + col] : xi[(long)b * 32 + (col - 96)];
  out[i] = (bf16_t)v;
}

// ---------------------------------------------------------------------------
// LayerNorm over H=512; writes f32 and bf16 copies. block=256, 2 elems/thread.
// ---------------------------------------------------------------------------
__global__ __launch_bounds__(256)
void layernorm_kernel(const float* __restrict__ in, const float* __restrict__ g,
                      const float* __restrict__ bvec, float* __restrict__ outf,
                      bf16_t* __restrict__ outb, int Hdim)
{
  const long row = blockIdx.x;
  const int tid = threadIdx.x;
  const float* xr = in + row * Hdim;
  float x0 = xr[tid], x1 = xr[tid + 256];

  __shared__ float ssum[256], ssq[256];
  ssum[tid] = x0 + x1;
  ssq[tid]  = x0 * x0 + x1 * x1;
  __syncthreads();
  for (int s = 128; s > 0; s >>= 1) {
    if (tid < s) { ssum[tid] += ssum[tid + s]; ssq[tid] += ssq[tid + s]; }
    __syncthreads();
  }
  float mean = ssum[0] / (float)Hdim;
  float var  = ssq[0] / (float)Hdim - mean * mean;
  float inv  = rsqrtf(var + 1e-5f);

  float y0 = (x0 - mean) * inv * g[tid]       + bvec[tid];
  float y1 = (x1 - mean) * inv * g[tid + 256] + bvec[tid + 256];
  outf[row * Hdim + tid]        = y0;
  outf[row * Hdim + tid + 256]  = y1;
  outb[row * Hdim + tid]        = (bf16_t)y0;
  outb[row * Hdim + tid + 256]  = (bf16_t)y1;
}

// ---------------------------------------------------------------------------
// Depthwise causal conv (k=4) + bias + SiLU. Reads u = xz[:, 0:1024] (ld 2048),
// writes f32 uc (ld 1024) and bf16 copy.
// ---------------------------------------------------------------------------
__global__ __launch_bounds__(256)
void conv_silu_kernel(const float* __restrict__ xz, const float* __restrict__ cw,
                      const float* __restrict__ cb, float* __restrict__ uc,
                      bf16_t* __restrict__ ub, int Tlen, long n)
{
  long i = (long)blockIdx.x * 256 + threadIdx.x;
  if (i >= n) return;
  int d = (int)(i & 1023);
  long row = i >> 10;
  int t = (int)(row & (long)(Tlen - 1));

  float acc = cb[d];
#pragma unroll
  for (int j = 0; j < 4; ++j) {
    int tt = t - 3 + j;
    if (tt >= 0) acc += xz[(row - (3 - j)) * 2048 + d] * cw[d * 4 + j];
  }
  float v = siluf_(acc);
  uc[i] = v;
  ub[i] = (bf16_t)v;
}

// ---------------------------------------------------------------------------
// Selective scan: block = 256 channels of one batch; B/C staged in LDS.
// h[s] = exp2(dt*A2[s])*h[s] + dt*u*B[s];  y = <h,C> + D*u
// (A2 folds log2(e) so the per-step transcendental is a raw v_exp_f32.)
// ys may alias delta (same-element read-before-write).
// ---------------------------------------------------------------------------
__global__ __launch_bounds__(256)
void scan_kernel(const float* __restrict__ delta, const float* __restrict__ uc,
                 const float* __restrict__ xdbl, const float* __restrict__ alog,
                 const float* __restrict__ Dp, float* __restrict__ ys, int Tlen)
{
  const int b = blockIdx.x;
  const int d = blockIdx.y * 256 + threadIdx.x;
  const float LOG2E = 1.4426950408889634f;

  float Ac[16];
#pragma unroll
  for (int s = 0; s < 16; ++s) Ac[s] = -__expf(alog[d * 16 + s]) * LOG2E;
  float h[16];
#pragma unroll
  for (int s = 0; s < 16; ++s) h[s] = 0.0f;
  const float Dd = Dp[d];

  __shared__ float sB[16], sC[16];

  for (int t = 0; t < Tlen; ++t) {
    __syncthreads();
    if (threadIdx.x < 32) {
      long row = (long)b * Tlen + t;
      float v = xdbl[row * 64 + 32 + threadIdx.x];
      if (threadIdx.x < 16) sB[threadIdx.x] = v;
      else                  sC[threadIdx.x - 16] = v;
    }
    __syncthreads();

    long idx = ((long)b * Tlen + t) * 1024 + d;
    float dt = delta[idx];
    float ut = uc[idx];
    float du = dt * ut;
    float y = 0.0f;
#pragma unroll
    for (int s = 0; s < 16; ++s) {
      h[s] = exp2f(dt * Ac[s]) * h[s] + du * sB[s];
      y += h[s] * sC[s];
    }
    ys[idx] = y + Dd * ut;
  }
}

// y *= silu(z) where z = xz[:, 1024:2048]; writes bf16
__global__ __launch_bounds__(256)
void gate_kernel(const float* __restrict__ ys, const float* __restrict__ xz,
                 bf16_t* __restrict__ out, long n)
{
  long i = (long)blockIdx.x * 256 + threadIdx.x;
  if (i >= n) return;
  int d = (int)(i & 1023);
  long row = i >> 10;
  float z = xz[row * 2048 + 1024 + d];
  out[i] = (bf16_t)(ys[i] * siluf_(z));
}

// ---------------------------------------------------------------------------
// Attention pooling + fc head. One block per batch element.
// ---------------------------------------------------------------------------
__global__ __launch_bounds__(256)
void pool_kernel(const float* __restrict__ x, const float* __restrict__ attn_w,
                 const float* __restrict__ attn_b, const float* __restrict__ fc_w,
                 const float* __restrict__ fc_b, float* __restrict__ out,
                 int Tlen, int Hdim)
{
  __shared__ float sw[1024];
  __shared__ float sred[256];
  const int b = blockIdx.x, tid = threadIdx.x;
  const float* xb = x + (size_t)b * Tlen * Hdim;

  for (int t = tid; t < Tlen; t += 256) {
    const float* xr = xb + (size_t)t * Hdim;
    float s = 0.0f;
    for (int hh = 0; hh < Hdim; ++hh) s += xr[hh] * attn_w[hh];
    sw[t] = s + attn_b[0];
  }
  __syncthreads();

  float mx = -3.4e38f;
  for (int t = tid; t < Tlen; t += 256) mx = fmaxf(mx, sw[t]);
  sred[tid] = mx; __syncthreads();
  for (int s = 128; s > 0; s >>= 1) { if (tid < s) sred[tid] = fmaxf(sred[tid], sred[tid + s]); __syncthreads(); }
  mx = sred[0]; __syncthreads();

  float psum = 0.0f;
  for (int t = tid; t < Tlen; t += 256) { float e = __expf(sw[t] - mx); sw[t] = e; psum += e; }
  sred[tid] = psum; __syncthreads();
  for (int s = 128; s > 0; s >>= 1) { if (tid < s) sred[tid] += sred[tid + s]; __syncthreads(); }
  float tot = sred[0]; __syncthreads();

  float acc = 0.0f;
  for (int hh = tid; hh < Hdim; hh += 256) {
    float p = 0.0f;
    for (int t = 0; t < Tlen; ++t) p += sw[t] * xb[(size_t)t * Hdim + hh];
    acc += p * fc_w[hh];
  }
  acc /= tot;
  sred[tid] = acc; __syncthreads();
  for (int s = 128; s > 0; s >>= 1) { if (tid < s) sred[tid] += sred[tid + s]; __syncthreads(); }
  if (tid == 0) out[b] = sred[0] + fc_b[0];
}

// ---------------------------------------------------------------------------
extern "C" void kernel_launch(void* const* d_in, const int* in_sizes, int n_in,
                              void* d_out, int out_size, void* d_ws, size_t ws_size,
                              hipStream_t stream)
{
  const int Bb = 16, Tt = 1024, Hh = 512, DINc = 1024, DTRc = 32, FIN = 128, XD = 64, Ll = 2;
  const long Mm = (long)Bb * Tt;                    // 16384 token rows

  const float* xv        = (const float*)d_in[0];
  const float* xi        = (const float*)d_in[1];
  const float* win_w     = (const float*)d_in[2];
  const float* win_b     = (const float*)d_in[3];
  const float* ln_in_g   = (const float*)d_in[4];
  const float* ln_in_b   = (const float*)d_in[5];
  const float* m_inproj  = (const float*)d_in[6];
  const float* m_convw   = (const float*)d_in[7];
  const float* m_convb   = (const float*)d_in[8];
  const float* m_xproj   = (const float*)d_in[9];
  const float* m_dtw     = (const float*)d_in[10];
  const float* m_dtb     = (const float*)d_in[11];
  const float* m_alog    = (const float*)d_in[12];
  const float* m_d       = (const float*)d_in[13];
  const float* m_outproj = (const float*)d_in[14];
  const float* blk_g     = (const float*)d_in[15];
  const float* blk_b     = (const float*)d_in[16];
  const float* attn_w    = (const float*)d_in[17];
  const float* attn_b    = (const float*)d_in[18];
  const float* fc_w      = (const float*)d_in[19];
  const float* fc_b      = (const float*)d_in[20];
  float* outp = (float*)d_out;

  // ---- workspace carve-up (256B aligned) ----
  char* ws = (char*)d_ws;
  size_t off = 0;
  auto carve = [&](size_t bytes) -> char* {
    char* p = ws + off;
    off = (off + bytes + 255) & ~(size_t)255;
    return p;
  };
  bf16_t* bf_win     = (bf16_t*)carve((size_t)Hh * FIN * 2);
  bf16_t* bf_inproj  = (bf16_t*)carve((size_t)Ll * 2 * DINc * Hh * 2);
  bf16_t* bf_xproj   = (bf16_t*)carve((size_t)Ll * XD * DINc * 2);
  bf16_t* bf_dtw     = (bf16_t*)carve((size_t)Ll * DINc * DTRc * 2);
  bf16_t* bf_outproj = (bf16_t*)carve((size_t)Ll * Hh * DINc * 2);
  bf16_t* bf_xcat    = (bf16_t*)carve((size_t)Mm * FIN * 2);
  float*  f_tmp      = (float*) carve((size_t)Mm * Hh * 4);
  float*  f_x        = (float*) carve((size_t)Mm * Hh * 4);
  bf16_t* bf_x       = (bf16_t*)carve((size_t)Mm * Hh * 2);
  float*  f_xz       = (float*) carve((size_t)Mm * 2 * DINc * 4);
  float*  f_uc       = (float*) carve((size_t)Mm * DINc * 4);
  bf16_t* bf_u       = (bf16_t*)carve((size_t)Mm * DINc * 2);   // reused as bf_y
  float*  f_xdbl     = (float*) carve((size_t)Mm * XD * 4);
  bf16_t* bf_dtr     = (bf16_t*)carve((size_t)Mm * DTRc * 2);
  float*  f_delta    = (float*) carve((size_t)Mm * DINc * 4);   // reused as ys
  if (off > ws_size) return;   // workspace too small -> no-op

  auto gemm = [&](const bf16_t* A, const bf16_t* W, float* C, const float* bias,
                  int M, int N, int K, int ldc, int act) {
    dim3 grid(N / 64, M / 128);
    if (act == 1)
      gemm_bf16_wmma<1><<<grid, dim3(128), 0, stream>>>(A, W, C, bias, M, N, K, ldc);
    else
      gemm_bf16_wmma<0><<<grid, dim3(128), 0, stream>>>(A, W, C, bias, M, N, K, ldc);
  };
  auto cast = [&](const float* in, bf16_t* out, long n) {
    cast_f32_bf16_kernel<<<(n + 255) / 256, 256, 0, stream>>>(in, out, n);
  };

  // weights -> bf16 (all layers at once)
  cast(win_w,     bf_win,     (long)Hh * FIN);
  cast(m_inproj,  bf_inproj,  (long)Ll * 2 * DINc * Hh);
  cast(m_xproj,   bf_xproj,   (long)Ll * XD * DINc);
  cast(m_dtw,     bf_dtw,     (long)Ll * DINc * DTRc);
  cast(m_outproj, bf_outproj, (long)Ll * Hh * DINc);

  // input fusion + projection + LN
  concat_cast_kernel<<<(Mm * FIN + 255) / 256, 256, 0, stream>>>(xv, xi, bf_xcat, Mm * FIN, Tt);
  gemm(bf_xcat, bf_win, f_tmp, win_b, (int)Mm, Hh, FIN, Hh, 0);
  layernorm_kernel<<<(int)Mm, 256, 0, stream>>>(f_tmp, ln_in_g, ln_in_b, f_x, bf_x, Hh);

  for (int l = 0; l < Ll; ++l) {
    const bf16_t* w_in  = bf_inproj  + (size_t)l * 2 * DINc * Hh;
    const bf16_t* w_xp  = bf_xproj   + (size_t)l * XD * DINc;
    const bf16_t* w_dt  = bf_dtw     + (size_t)l * DINc * DTRc;
    const bf16_t* w_out = bf_outproj + (size_t)l * Hh * DINc;
    const float* cw   = m_convw + (size_t)l * DINc * 4;
    const float* cb   = m_convb + (size_t)l * DINc;
    const float* dtb  = m_dtb   + (size_t)l * DINc;
    const float* alog = m_alog  + (size_t)l * DINc * 16;
    const float* Dp   = m_d     + (size_t)l * DINc;

    // xz = x @ inproj^T  (M, 2048)
    gemm(bf_x, w_in, f_xz, nullptr, (int)Mm, 2 * DINc, Hh, 2 * DINc, 0);
    // u = silu(causal depthwise conv(u) + b)
    conv_silu_kernel<<<(Mm * DINc + 255) / 256, 256, 0, stream>>>(f_xz, cw, cb, f_uc, bf_u, Tt, Mm * DINc);
    // xdbl = u @ xproj^T  (M, 64)
    gemm(bf_u, w_xp, f_xdbl, nullptr, (int)Mm, XD, DINc, XD, 0);
    // delta = softplus(dtr @ dtw^T + dtb)  (M, 1024)
    cast_strided_kernel<<<(Mm * DTRc + 255) / 256, 256, 0, stream>>>(f_xdbl, XD, DTRc, bf_dtr, Mm * DTRc);
    gemm(bf_dtr, w_dt, f_delta, dtb, (int)Mm, DINc, DTRc, DINc, 1);
    // selective scan (ys aliases f_delta)
    scan_kernel<<<dim3(Bb, DINc / 256), 256, 0, stream>>>(f_delta, f_uc, f_xdbl, alog, Dp, f_delta, Tt);
    // y *= silu(z) -> bf16 (aliases bf_u)
    gate_kernel<<<(Mm * DINc + 255) / 256, 256, 0, stream>>>(f_delta, f_xz, bf_u, Mm * DINc);
    // out = y @ outproj^T, then LN -> next x
    gemm(bf_u, w_out, f_tmp, nullptr, (int)Mm, Hh, DINc, Hh, 0);
    layernorm_kernel<<<(int)Mm, 256, 0, stream>>>(f_tmp, blk_g + (size_t)l * Hh, blk_b + (size_t)l * Hh,
                                                  f_x, bf_x, Hh);
  }

  // attention pooling + fc
  pool_kernel<<<Bb, 256, 0, stream>>>(f_x, attn_w, attn_b, fc_w, fc_b, outp, Tt, Hh);
}
